// ScaledDotProductAttention_36867999269479
// MI455X (gfx1250) — compile-verified, matrix operand order checked by reading, not run on hardware
//
#include <hip/hip_runtime.h>
#include <hip/hip_bf16.h>
#include <math.h>

typedef __attribute__((ext_vector_type(16))) __bf16 v16bf;
typedef __attribute__((ext_vector_type(8)))  float  v8f;
typedef __attribute__((ext_vector_type(4)))  unsigned int u32x4;
typedef __attribute__((ext_vector_type(8)))  int i32x8;
typedef __attribute__((ext_vector_type(4)))  int i32x4;

#define B_  32
#define LQ_ 2048
#define LK_ 2048
#define DK_ 64
#define QTILE 64      // q rows per workgroup
#define KTILE 128     // keys per inner tile
#define NTHREADS 256  // 8 waves (wave32)

// K-index packing inside a 16-bit A/B fragment VGPR group:
// VGPR v (0..7), half h (lane>=16): K = (v/4)*16 + h*8 + (v%4)*2  (pair K,K+1)
__device__ __forceinline__ int kidx(int v, int h) {
    return ((v >> 2) << 4) + (h << 3) + ((v & 3) << 1);
}

// A-fragment (16x32 bf16, row-major source in LDS, stride elements)
__device__ __forceinline__ v16bf load_afrag(const __bf16* t, int rowBase, int colBase,
                                            int stride, int lane) {
    const int m = lane & 15, h = lane >> 4;
    v16bf f;
#pragma unroll
    for (int v = 0; v < 8; ++v) {
        const int K = colBase + kidx(v, h);
        const int a = (rowBase + m) * stride + K;
        f[2 * v]     = t[a];
        f[2 * v + 1] = t[a + 1];
    }
    return f;
}

// B-fragment for S = Q*K^T : B[Kdim=d][n=key], source ktile[key][d] (d pairs contiguous)
__device__ __forceinline__ v16bf load_bfrag_kT(const __bf16* t, int keyBase, int dBase,
                                               int lane) {
    const int n = lane & 15, h = lane >> 4;
    v16bf f;
#pragma unroll
    for (int v = 0; v < 8; ++v) {
        const int d = dBase + kidx(v, h);
        const int a = (keyBase + n) * DK_ + d;
        f[2 * v]     = t[a];
        f[2 * v + 1] = t[a + 1];
    }
    return f;
}

// B-fragment for O = P*V : B[Kdim=key][n=dcol], source vtile[key][d] (pair strides a row)
__device__ __forceinline__ v16bf load_bfrag_v(const __bf16* t, int kBase, int colBase,
                                              int lane) {
    const int n = lane & 15, h = lane >> 4;
    v16bf f;
#pragma unroll
    for (int v = 0; v < 8; ++v) {
        const int k = kBase + kidx(v, h);
        f[2 * v]     = t[k * DK_ + colBase + n];
        f[2 * v + 1] = t[(k + 1) * DK_ + colBase + n];
    }
    return f;
}

// cooperative fp32 global -> bf16 LDS tile copy (nElems multiple of 1024)
__device__ __forceinline__ void load_tile_bf16(const float* __restrict__ src,
                                               __bf16* dst, int nElems, int tid) {
    for (int i = tid * 4; i < nElems; i += NTHREADS * 4) {
        const float4 x = *(const float4*)(src + i);
        dst[i + 0] = (__bf16)x.x;
        dst[i + 1] = (__bf16)x.y;
        dst[i + 2] = (__bf16)x.z;
        dst[i + 3] = (__bf16)x.w;
    }
}

// Stage the 64x128 byte mask tile (row stride LK_) into LDS.
// Preferred path: Tensor Data Mover (one async DMA, TENSORcnt-tracked), issued
// by wave 0 only (TDM ignores EXEC; one instruction per issuing wave).
// Fallback: coalesced uint4 copy.
__device__ __forceinline__ void stage_mask_tile(const unsigned char* __restrict__ gsrc,
                                                unsigned char* lds, int tid, int wave) {
#if defined(__has_builtin) && __has_builtin(__builtin_amdgcn_tensor_load_to_lds) && \
    __has_builtin(__builtin_amdgcn_s_wait_tensorcnt)
    if (wave == 0) {
        const unsigned long long ga = (unsigned long long)(uintptr_t)gsrc;
        const unsigned lds_off = (unsigned)(uintptr_t)lds;  // generic LDS addr: low 32b = LDS byte offset
        // D# group 0: count=1 | lds_addr | global_addr[56:0] | type=2
        u32x4 g0;
        g0[0] = 1u;                                    // count=1, is_restore=0, gather=0
        g0[1] = lds_off;                               // lds_addr (bytes)
        g0[2] = (unsigned)(ga & 0xffffffffu);          // global_addr[31:0]
        g0[3] = (unsigned)((ga >> 32) & 0x01ffffffu) | 0x80000000u;  // ga[56:32] | type=2<<30
        // D# group 1: data_size=1B, tensor_dim0=LK, tensor_dim1=64,
        //             tile_dim0=128, tile_dim1=64, tensor_dim0_stride=LK
        i32x8 g1;
        g1[0] = 0;                       // wg_mask=0, data_size=0(1B), no flags
        g1[1] = (int)(2048u << 16);      // tensor_dim0[15:0]=2048 at bits 63:48
        g1[2] = (int)(64u << 16);        // tensor_dim0[31:16]=0 | tensor_dim1[15:0]=64
        g1[3] = (int)(128u << 16);       // tensor_dim1[31:16]=0 | tile_dim0=128
        g1[4] = 64;                      // tile_dim1=64 | tile_dim2=0
        g1[5] = 2048;                    // tensor_dim0_stride[31:0]=2048
        g1[6] = 0;                       // stride[47:32]=0 | dim1_stride[15:0]=0
        g1[7] = 0;                       // dim1_stride[47:16]=0
        i32x4 g2 = {0, 0, 0, 0};
        i32x4 g3 = {0, 0, 0, 0};
#if __clang_major__ >= 23
        i32x8 g4 = {0, 0, 0, 0, 0, 0, 0, 0};
        __builtin_amdgcn_tensor_load_to_lds(g0, g1, g2, g3, g4, 0);
#else
        __builtin_amdgcn_tensor_load_to_lds(g0, g1, g2, g3, 0);
#endif
        __builtin_amdgcn_s_wait_tensorcnt(0);
    }
#else
    // 64 rows x 128B = 512 x 16B chunks; 2 per thread, consecutive tids walk a row
    for (int c = tid; c < (QTILE * KTILE) / 16; c += NTHREADS) {
        const int row = c >> 3, col16 = (c & 7) * 16;
        *(uint4*)(lds + row * KTILE + col16) =
            *(const uint4*)(gsrc + (size_t)row * LK_ + col16);
    }
#endif
}

__global__ __launch_bounds__(NTHREADS)
void attn_fwd_kernel(const float* __restrict__ q, const float* __restrict__ k,
                     const float* __restrict__ v, const unsigned char* __restrict__ mask,
                     float* __restrict__ out, float* __restrict__ p) {
    __shared__ __bf16 qtile[QTILE * DK_];            // 8 KB
    __shared__ __bf16 kvtile[KTILE * DK_];           // 16 KB (K in pass1, V in pass2)
    __shared__ unsigned char mtile[QTILE * KTILE];   // 8 KB mask tile
    __shared__ float  part[4][QTILE];                // per-wave_n row partials
    __shared__ float  m_s[QTILE], l_s[QTILE], corr_s[QTILE];

    const int tid  = threadIdx.x;
    const int lane = tid & 31;
    const int wave = tid >> 5;
    const int wm   = wave >> 2;   // 0..1 : 32 q-rows each
    const int wn   = wave & 3;    // 0..3 : key-cols (pass1) / O-cols (pass2)
    const int h    = lane >> 4;
    const int ln   = lane & 15;

    const int b  = blockIdx.y;
    const int q0 = blockIdx.x * QTILE;
    const float scaling = 0.125f; // 1/sqrt(64)

    const float* qb = q + ((size_t)b * LQ_ + q0) * DK_;
    const float* kb = k + (size_t)b * LK_ * DK_;
    const float* vb = v + (size_t)b * LK_ * DK_;
    const unsigned char* mb = mask + ((size_t)b * LQ_ + q0) * LK_;

    if (tid < QTILE) { m_s[tid] = -INFINITY; l_s[tid] = 0.0f; }
    load_tile_bf16(qb, qtile, QTILE * DK_, tid);
    __syncthreads();

    // Q A-fragments, resident in registers for the whole kernel.
    v16bf qf[2][2];
#pragma unroll
    for (int mi = 0; mi < 2; ++mi)
#pragma unroll
        for (int kk = 0; kk < 2; ++kk)
            qf[mi][kk] = load_afrag(qtile, wm * 32 + mi * 16, kk * 32, DK_, lane);

    // ---------------- Pass 1: S = scale*Q*K^T -> p buffer; online row max/sum ---------
    for (int kt = 0; kt < LK_ / KTILE; ++kt) {
        const int k0 = kt * KTILE;
        stage_mask_tile(mb + k0, mtile, tid, wave);              // async TDM (wave 0)
        load_tile_bf16(kb + (size_t)k0 * DK_, kvtile, KTILE * DK_, tid);
        if (kt + 1 < LK_ / KTILE)                                // prefetch next K tile
            __builtin_prefetch(kb + (size_t)(k0 + KTILE) * DK_ + tid * 32, 0, 1);
        __syncthreads();

        v8f accS[2][2];
#pragma unroll
        for (int mi = 0; mi < 2; ++mi) {
#pragma unroll
            for (int ni = 0; ni < 2; ++ni) {
                v8f acc = {0.f, 0.f, 0.f, 0.f, 0.f, 0.f, 0.f, 0.f};
#pragma unroll
                for (int kk = 0; kk < 2; ++kk) {
                    const v16bf bf = load_bfrag_kT(kvtile, wn * 32 + ni * 16, kk * 32, lane);
                    acc = __builtin_amdgcn_wmma_f32_16x16x32_bf16(
                        false, qf[mi][kk], false, bf, (short)0, acc, false, false);
                }
                // scale, mask (from LDS tile), stage raw S into the p output buffer
                const int klocal = wn * 32 + ni * 16 + ln;
                const int kcol   = k0 + klocal;
#pragma unroll
                for (int r = 0; r < 8; ++r) {
                    const int rlocal = wm * 32 + mi * 16 + r + 8 * h;
                    float val = acc[r] * scaling;
                    if (mtile[rlocal * KTILE + klocal]) val = -INFINITY;
                    acc[r] = val;
                    p[((size_t)b * LQ_ + q0 + rlocal) * (size_t)LK_ + kcol] = val;
                }
                accS[mi][ni] = acc;
            }
            // per-row tile max: across ni, then across the 16 lanes of each half
#pragma unroll
            for (int r = 0; r < 8; ++r) {
                float tm = fmaxf(accS[mi][0][r], accS[mi][1][r]);
#pragma unroll
                for (int off = 8; off >= 1; off >>= 1)
                    tm = fmaxf(tm, __shfl_xor(tm, off, 16));
                if (ln == 0) part[wn][wm * 32 + mi * 16 + r + 8 * h] = tm;
            }
        }
        __syncthreads();

        if (tid < QTILE) {
            float tm = fmaxf(fmaxf(part[0][tid], part[1][tid]),
                             fmaxf(part[2][tid], part[3][tid]));
            const float mo = m_s[tid];
            const float mn = fmaxf(mo, tm);
            m_s[tid]    = mn;
            corr_s[tid] = __expf(mo - mn);
        }
        __syncthreads();

        // per-row sum of exp(s - m_new)
#pragma unroll
        for (int mi = 0; mi < 2; ++mi) {
#pragma unroll
            for (int r = 0; r < 8; ++r) {
                const int row = wm * 32 + mi * 16 + r + 8 * h;
                const float mn = m_s[row];
                float s = __expf(accS[mi][0][r] - mn) + __expf(accS[mi][1][r] - mn);
#pragma unroll
                for (int off = 8; off >= 1; off >>= 1)
                    s += __shfl_xor(s, off, 16);
                if (ln == 0) part[wn][row] = s;
            }
        }
        __syncthreads();

        if (tid < QTILE)
            l_s[tid] = l_s[tid] * corr_s[tid] +
                       part[0][tid] + part[1][tid] + part[2][tid] + part[3][tid];
        __syncthreads();
    }

    if (tid < QTILE) l_s[tid] = 1.0f / l_s[tid];  // reciprocal sum for normalization
    __threadfence();  // make staged S globally visible to all waves of this group
    __syncthreads();

    // ---------------- Pass 2: p = exp(s-m)/l (write back), O += P*V ---------------------
    v8f accO[2] = {{0.f, 0.f, 0.f, 0.f, 0.f, 0.f, 0.f, 0.f},
                   {0.f, 0.f, 0.f, 0.f, 0.f, 0.f, 0.f, 0.f}};

    for (int kt = 0; kt < LK_ / KTILE; ++kt) {
        const int k0 = kt * KTILE;
        __syncthreads();  // previous iteration done reading kvtile
        load_tile_bf16(vb + (size_t)k0 * DK_, kvtile, KTILE * DK_, tid);
        if (kt + 1 < LK_ / KTILE)                                // prefetch next V tile
            __builtin_prefetch(vb + (size_t)(k0 + KTILE) * DK_ + tid * 32, 0, 1);
        __syncthreads();

        // V B-fragments for this key tile (shared across both M subtiles)
        v16bf vf[4];
#pragma unroll
        for (int kk = 0; kk < 4; ++kk)
            vf[kk] = load_bfrag_v(kvtile, kk * 32, wn * 16, lane);

#pragma unroll
        for (int mi = 0; mi < 2; ++mi) {
            const int rowl = wm * 32 + mi * 16 + ln;
            const float mrow = m_s[rowl];
            const float rl   = l_s[rowl];
            const size_t srow = ((size_t)b * LQ_ + q0 + rowl) * (size_t)LK_ + k0;
#pragma unroll
            for (int kk = 0; kk < 4; ++kk) {
                v16bf pf;
#pragma unroll
                for (int vv = 0; vv < 8; ++vv) {
                    const int K = kk * 32 + kidx(vv, h);
                    float2 s2 = *(float2*)(p + srow + K);
                    const float e0 = __expf(s2.x - mrow) * rl;
                    const float e1 = __expf(s2.y - mrow) * rl;
                    s2.x = e0; s2.y = e1;
                    *(float2*)(p + srow + K) = s2;  // normalized probabilities out
                    pf[2 * vv]     = (__bf16)e0;
                    pf[2 * vv + 1] = (__bf16)e1;
                }
                accO[mi] = __builtin_amdgcn_wmma_f32_16x16x32_bf16(
                    false, pf, false, vf[kk], (short)0, accO[mi], false, false);
            }
        }
    }

    // write O (each (row,col) owned by exactly one wave/lane/vgpr)
#pragma unroll
    for (int mi = 0; mi < 2; ++mi) {
#pragma unroll
        for (int r = 0; r < 8; ++r) {
            const int qrow = q0 + wm * 32 + mi * 16 + r + 8 * h;
            out[((size_t)b * LQ_ + qrow) * DK_ + wn * 16 + ln] = accO[mi][r];
        }
    }
}

extern "C" void kernel_launch(void* const* d_in, const int* in_sizes, int n_in,
                              void* d_out, int out_size, void* d_ws, size_t ws_size,
                              hipStream_t stream) {
    (void)in_sizes; (void)n_in; (void)out_size; (void)d_ws; (void)ws_size;
    const float* q = (const float*)d_in[0];
    const float* k = (const float*)d_in[1];
    const float* v = (const float*)d_in[2];
    const unsigned char* mask = (const unsigned char*)d_in[3];

    float* out = (float*)d_out;                                   // [B, LQ, DK]
    float* p   = out + (size_t)B_ * LQ_ * DK_;                    // [B, LQ, LK]

    dim3 grid(LQ_ / QTILE, B_);
    dim3 block(NTHREADS);
    attn_fwd_kernel<<<grid, block, 0, stream>>>(q, k, v, mask, out, p);
}
